// Net_58033598104031
// MI455X (gfx1250) — compile-verified, matrix operand order checked by reading, not run on hardware
//
#include <hip/hip_runtime.h>
#include <hip/hip_bf16.h>
#include <cstdint>
#include <cstddef>

// ---------------------------------------------------------------------------
// DGCNN (DynamicEdgeConv x4 + TopKPooling + classifier) for MI455X (gfx1250).
// Edge MLPs run as v_wmma_f32_16x16x32_f16 on LDS-resident per-node-block
// tiles; max-over-K aggregation via atomicMax(uint) on post-ReLU (>=0) data.
// ---------------------------------------------------------------------------

typedef __attribute__((ext_vector_type(16))) _Float16 v16h;
typedef __attribute__((ext_vector_type(8)))  float    v8f;
typedef __attribute__((ext_vector_type(4)))  float    v4f;

#define BATCH    16
#define KNN      20
#define NODE_BLK 4
#define RROWS    (NODE_BLK * KNN)   // 80 edge rows per workgroup
#define MTILES   (RROWS / 16)       // 5 WMMA M-tiles

static_assert(RROWS % 16 == 0, "rows must tile by 16");

union V16U { v4f f4[2]; v16h h; };

// Load one lane's A/B operand slice: halves [kb..kb+7] and [16+kb..16+kb+7]
// of a row-major f16 row -> two 16B loads (matches ISA 16-bit A 16x32 layout).
__device__ __forceinline__ v16h load_mat16(const _Float16* base) {
  V16U u;
  const v4f* p = reinterpret_cast<const v4f*>(base);
  u.f4[0] = p[0];   // halves +0..+7
  u.f4[1] = p[2];   // halves +16..+23
  return u.h;
}

// One MLP layer over the workgroup's 80xKIN LDS tile: D = relu(A*W^T + bias).
// If LAST, max-aggregate over KNN rows straight into xout via atomicMax.
template<int KIN, int COUT, bool LAST>
__device__ __forceinline__ void mlp_layer(
    const _Float16* __restrict__ inbuf, int sIn,
    const _Float16* __restrict__ wt,    const float* __restrict__ bias,
    _Float16* __restrict__ outbuf, int sOut,
    float* __restrict__ xout, int nodeBase, int ncolStride)
{
  const int tid   = threadIdx.x;
  const int wave  = tid >> 5;
  const int lane  = tid & 31;
  const int l16   = lane & 15;
  const int hi    = (lane >> 4) & 1;
  const int kbase = hi ? 8 : 0;
  constexpr int NT     = COUT / 16;
  constexpr int KSTEPS = KIN / 32;

  for (int job = wave; job < MTILES * NT; job += 4) {
    const int mt = job % MTILES;
    const int nt = job / MTILES;
    const float bval = bias[nt * 16 + l16];
    v8f acc;
#pragma unroll
    for (int r = 0; r < 8; ++r) acc[r] = bval;

    const _Float16* arow = inbuf + (size_t)(mt * 16 + l16) * sIn + kbase;
    const _Float16* brow = wt    + (size_t)(nt * 16 + l16) * KIN + kbase;
#pragma unroll
    for (int ks = 0; ks < KSTEPS; ++ks) {
      v16h av = load_mat16(arow + ks * 32);
      v16h bv = load_mat16(brow + ks * 32);
      acc = __builtin_amdgcn_wmma_f32_16x16x32_f16(
          false, av, false, bv, (short)0, acc, false, false);
    }

    if constexpr (!LAST) {
#pragma unroll
      for (int r = 0; r < 8; ++r) {
        float v = fmaxf(acc[r], 0.f);
        outbuf[(size_t)(mt * 16 + hi * 8 + r) * sOut + nt * 16 + l16] = (_Float16)v;
      }
    } else {
#pragma unroll
      for (int r = 0; r < 8; ++r) {
        float v = fmaxf(acc[r], 0.f);
        const int rowl = mt * 16 + hi * 8 + r;
        const int node = nodeBase + rowl / KNN;
        const int col  = nt * 16 + l16;
        atomicMax(reinterpret_cast<unsigned int*>(&xout[(size_t)node * ncolStride + col]),
                  __float_as_uint(v));   // valid: v >= 0, dst zero-initialized
      }
    }
  }
}

// Fused DynamicEdgeConv: gather [x_i, x_j-x_i] -> LDS f16, run NL MLP layers
// with WMMA, max over K neighbors into xout (zero-initialized).
template<int K1, int NL, int CLAST>
__global__ void __launch_bounds__(128) edge_conv_kernel(
    const float* __restrict__ x, const int* __restrict__ nbr,
    const _Float16* __restrict__ wt1, const float* __restrict__ b1,
    const _Float16* __restrict__ wt2, const float* __restrict__ b2,
    const _Float16* __restrict__ wt3, const float* __restrict__ b3,
    float* __restrict__ xout, int N, int Cact)
{
  constexpr int SA = K1 + 8;   // padded LDS strides (halves), 16B aligned
  constexpr int S2 = 72;
  constexpr int SM = RROWS * SA + RROWS * S2 + (NL == 3 ? RROWS * S2 : 0);
  __shared__ __align__(16) _Float16 smem[SM];
  _Float16* p0 = smem;
  _Float16* p1 = smem + RROWS * SA;
  _Float16* p2 = (NL == 3) ? (p1 + RROWS * S2) : p1;

  const int b        = blockIdx.y;
  const int nodeBase = blockIdx.x * NODE_BLK;
  const int tid      = threadIdx.x;
  const float* xb = x + (size_t)b * N * Cact;
  const int*   nb = nbr + ((size_t)b * N + nodeBase) * KNN;

  // Gather edge features [x_i | x_j - x_i | zero-pad] into LDS (f16).
  for (int e = tid; e < RROWS * K1; e += 128) {
    const int row  = e / K1, c = e % K1;
    const int nloc = row / KNN, kk = row % KNN;
    const int i = nodeBase + nloc;
    const int j = nb[nloc * KNN + kk];
    float v = 0.f;
    if (c < Cact)            v = xb[(size_t)i * Cact + c];
    else if (c < 2 * Cact) {
      const int cc = c - Cact;
      v = xb[(size_t)j * Cact + cc] - xb[(size_t)i * Cact + cc];
    }
    p0[(size_t)row * SA + c] = (_Float16)v;
  }
  __syncthreads();

  float* xoutb = xout + (size_t)b * N * CLAST;
  mlp_layer<K1, 64, false>(p0, SA, wt1, b1, p1, S2, nullptr, 0, 0);
  __syncthreads();
  if constexpr (NL == 3) {
    mlp_layer<64, 64, false>(p1, S2, wt2, b2, p2, S2, nullptr, 0, 0);
    __syncthreads();
    mlp_layer<64, CLAST, true>(p2, S2, wt3, b3, nullptr, 0, xoutb, nodeBase, CLAST);
  } else {
    mlp_layer<64, CLAST, true>(p1, S2, wt2, b2, nullptr, 0, xoutb, nodeBase, CLAST);
  }
}

// Fused kNN: thread-per-node, 64-candidate LDS tiles, register-resident
// fully-unrolled top-20 insertion list. Includes self at distance 0.
template<int C>
__global__ void __launch_bounds__(128) knn_kernel(
    const float* __restrict__ x, int* __restrict__ nbr, int N)
{
  __shared__ float xt[64 * C];
  const int b = blockIdx.y;
  const int i = blockIdx.x * 128 + threadIdx.x;
  const float* xb = x + (size_t)b * N * C;

  float xi[C];
#pragma unroll
  for (int c = 0; c < C; ++c) xi[c] = xb[(size_t)i * C + c];

  float bd[KNN]; int bi[KNN];
#pragma unroll
  for (int t = 0; t < KNN; ++t) { bd[t] = 3.4e38f; bi[t] = 0; }

  for (int j0 = 0; j0 < N; j0 += 64) {
    __syncthreads();
    for (int e = threadIdx.x; e < 64 * C; e += 128)
      xt[e] = xb[(size_t)j0 * C + e];
    __syncthreads();
    if (j0 + 64 < N)
      __builtin_prefetch(xb + (size_t)(j0 + 64) * C, 0, 1);  // global_prefetch_b8

    for (int jj = 0; jj < 64; ++jj) {
      float d = 0.f;
#pragma unroll
      for (int c = 0; c < C; ++c) {
        const float t = xi[c] - xt[jj * C + c];
        d = fmaf(t, t, d);
      }
      if (d < bd[KNN - 1]) {
        bd[KNN - 1] = d; bi[KNN - 1] = j0 + jj;
#pragma unroll
        for (int t = KNN - 1; t > 0; --t) {
          if (bd[t] < bd[t - 1]) {
            const float td = bd[t]; bd[t] = bd[t - 1]; bd[t - 1] = td;
            const int   ti = bi[t]; bi[t] = bi[t - 1]; bi[t - 1] = ti;
          }
        }
      }
    }
  }
  int* out = nbr + ((size_t)b * N + i) * KNN;
#pragma unroll
  for (int t = 0; t < KNN; ++t) out[t] = bi[t];
}

// TopKPooling: score = tanh(x.p/||p||); select top N/2 via in-LDS threshold
// bisection (output order irrelevant to final network output), scale by score.
__global__ void __launch_bounds__(256) topk_pool_kernel(
    const float* __restrict__ x, const float* __restrict__ p,
    float* __restrict__ xout, int N)
{
  constexpr int C = 64;
  __shared__ float s[2048];
  __shared__ float red[256];
  __shared__ int   ired[256];
  __shared__ int   cnt;
  const int b = blockIdx.x, tid = threadIdx.x;
  const float* xb = x + (size_t)b * N * C;

  float a = 0.f;
  for (int c = tid; c < C; c += 256) a += p[c] * p[c];
  red[tid] = a; __syncthreads();
  for (int st = 128; st > 0; st >>= 1) {
    if (tid < st) red[tid] += red[tid + st];
    __syncthreads();
  }
  const float inv = rsqrtf(red[0]);

  for (int i = tid; i < N; i += 256) {
    float dot = 0.f;
    for (int c = 0; c < C; ++c) dot = fmaf(xb[(size_t)i * C + c], p[c], dot);
    s[i] = tanhf(dot * inv);
  }
  __syncthreads();

  const int k = N >> 1;
  float lo = -1.001f, hi = 1.001f;
  for (int it = 0; it < 30; ++it) {
    const float mid = 0.5f * (lo + hi);
    int c = 0;
    for (int i = tid; i < N; i += 256) c += (s[i] >= mid) ? 1 : 0;
    ired[tid] = c; __syncthreads();
    for (int st = 128; st > 0; st >>= 1) {
      if (tid < st) ired[tid] += ired[tid + st];
      __syncthreads();
    }
    const int total = ired[0];
    __syncthreads();
    if (total >= k) lo = mid; else hi = mid;
  }

  if (tid == 0) cnt = 0;
  __syncthreads();
  float* ob = xout + (size_t)b * k * C;
  for (int i = tid; i < N; i += 256) {
    if (s[i] >= hi) {                       // count(>=hi) < k guaranteed
      const int pos = atomicAdd(&cnt, 1);
      const float sc = s[i];
      for (int c = 0; c < C; ++c) ob[(size_t)pos * C + c] = xb[(size_t)i * C + c] * sc;
    }
  }
  __syncthreads();
  for (int i = tid; i < N; i += 256) {
    if (s[i] >= lo && s[i] < hi) {          // fill remainder at the threshold
      const int pos = atomicAdd(&cnt, 1);
      if (pos < k) {
        const float sc = s[i];
        for (int c = 0; c < C; ++c) ob[(size_t)pos * C + c] = xb[(size_t)i * C + c] * sc;
      }
    }
  }
}

// Global max pool over nodes (inputs are post-ReLU, >= 0).
__global__ void gmax_kernel(const float* __restrict__ x, float* __restrict__ dst,
                            int N, int C, int colOff)
{
  const int b = blockIdx.y;
  const int c = blockIdx.x * 256 + threadIdx.x;
  if (c >= C) return;
  const float* xb = x + (size_t)b * N * C + c;
  float m = 0.f;
  for (int n = 0; n < N; ++n) m = fmaxf(m, xb[(size_t)n * C]);
  dst[b * 704 + colOff + c] = m;
}

__global__ void concat_kernel(const float* __restrict__ pos, const float* __restrict__ nrm,
                              float* __restrict__ x0, int total)
{
  const int g = blockIdx.x * 256 + threadIdx.x;
  if (g >= total) return;
#pragma unroll
  for (int c = 0; c < 3; ++c) {
    x0[(size_t)g * 6 + c]     = pos[(size_t)g * 3 + c];
    x0[(size_t)g * 6 + 3 + c] = nrm[(size_t)g * 3 + c];
  }
}

// W [Kin][Cout] f32 -> Wt [Cout][Kpad] f16 (transposed, zero-padded K).
__global__ void prep_wt_kernel(const float* __restrict__ W, _Float16* __restrict__ wt,
                               int Kin, int Kpad, int Cout)
{
  const int g = blockIdx.x * 256 + threadIdx.x;
  if (g >= Kpad * Cout) return;
  const int n = g / Kpad, k = g % Kpad;
  wt[g] = (k < Kin) ? (_Float16)W[(size_t)k * Cout + n] : (_Float16)0.f;
}

__global__ void fill_kernel(float* __restrict__ p, int n)
{
  const int g = blockIdx.x * 256 + threadIdx.x;
  if (g < n) p[g] = 0.f;
}

__global__ void linear_kernel(const float* __restrict__ in, const float* __restrict__ W,
                              const float* __restrict__ bias, float* __restrict__ out,
                              int rows, int Cin, int Cout, int doRelu)
{
  const int g = blockIdx.x * 256 + threadIdx.x;
  if (g >= rows * Cout) return;
  const int r = g / Cout, c = g % Cout;
  float acc = bias[c];
  const float* ir = in + (size_t)r * Cin;
  for (int k = 0; k < Cin; ++k) acc = fmaf(ir[k], W[(size_t)k * Cout + c], acc);
  out[g] = doRelu ? fmaxf(acc, 0.f) : acc;
}

__global__ void logsoftmax_kernel(const float* __restrict__ logits, float* __restrict__ out)
{
  const int b = threadIdx.x;
  if (b >= BATCH) return;
  const float* r = logits + b * 40;
  float m = -3.4e38f;
  for (int c = 0; c < 40; ++c) m = fmaxf(m, r[c]);
  float sum = 0.f;
  for (int c = 0; c < 40; ++c) sum += expf(r[c] - m);
  const float ls = logf(sum);
  for (int c = 0; c < 40; ++c) out[b * 40 + c] = r[c] - m - ls;
}

extern "C" void kernel_launch(void* const* d_in, const int* in_sizes, int n_in,
                              void* d_out, int out_size, void* d_ws, size_t ws_size,
                              hipStream_t stream)
{
  (void)in_sizes; (void)n_in; (void)out_size; (void)ws_size;
  const float* pos  = (const float*)d_in[0];
  const float* nrm  = (const float*)d_in[1];
  const float* c1W1 = (const float*)d_in[2];  const float* c1b1 = (const float*)d_in[3];
  const float* c1W2 = (const float*)d_in[4];  const float* c1b2 = (const float*)d_in[5];
  const float* c1W3 = (const float*)d_in[6];  const float* c1b3 = (const float*)d_in[7];
  const float* c2W1 = (const float*)d_in[8];  const float* c2b1 = (const float*)d_in[9];
  const float* c2W2 = (const float*)d_in[10]; const float* c2b2 = (const float*)d_in[11];
  const float* c3W1 = (const float*)d_in[12]; const float* c3b1 = (const float*)d_in[13];
  const float* c3W2 = (const float*)d_in[14]; const float* c3b2 = (const float*)d_in[15];
  const float* c4W1 = (const float*)d_in[16]; const float* c4b1 = (const float*)d_in[17];
  const float* c4W2 = (const float*)d_in[18]; const float* c4b2 = (const float*)d_in[19];
  const float* p1   = (const float*)d_in[20];
  const float* p2   = (const float*)d_in[21];
  const float* p3   = (const float*)d_in[22];
  const float* m1W  = (const float*)d_in[23]; const float* m1b = (const float*)d_in[24];
  const float* m2W  = (const float*)d_in[25]; const float* m2b = (const float*)d_in[26];
  const float* m3W  = (const float*)d_in[27]; const float* m3b = (const float*)d_in[28];
  float* dout = (float*)d_out;

  // ---- workspace layout (~34 MB) ----
  char* base = (char*)d_ws;
  size_t off = 0;
  auto alloc = [&](size_t bytes) -> void* {
    void* p = base + off;
    off += (bytes + 255) & ~(size_t)255;
    return p;
  };
  float* x0   = (float*)alloc(16ull * 2048 * 6   * 4);
  float* x1   = (float*)alloc(16ull * 2048 * 64  * 4);
  float* x1p  = (float*)alloc(16ull * 1024 * 64  * 4);
  float* x2   = (float*)alloc(16ull * 1024 * 64  * 4);
  float* x2p  = (float*)alloc(16ull * 512  * 64  * 4);
  float* x3   = (float*)alloc(16ull * 512  * 64  * 4);
  float* x3p  = (float*)alloc(16ull * 256  * 64  * 4);
  float* x4   = (float*)alloc(16ull * 256  * 512 * 4);
  int*   nbr  = (int*)  alloc(16ull * 2048 * KNN * 4);
  float* f704 = (float*)alloc(16ull * 704 * 4);
  float* h512 = (float*)alloc(16ull * 512 * 4);
  float* h256 = (float*)alloc(16ull * 256 * 4);
  float* lgt  = (float*)alloc(16ull * 40  * 4);
  _Float16* w11 = (_Float16*)alloc(64ull  * 32  * 2);
  _Float16* w12 = (_Float16*)alloc(64ull  * 64  * 2);
  _Float16* w13 = (_Float16*)alloc(64ull  * 64  * 2);
  _Float16* w21 = (_Float16*)alloc(64ull  * 128 * 2);
  _Float16* w22 = (_Float16*)alloc(64ull  * 64  * 2);
  _Float16* w31 = (_Float16*)alloc(64ull  * 128 * 2);
  _Float16* w32 = (_Float16*)alloc(64ull  * 64  * 2);
  _Float16* w41 = (_Float16*)alloc(64ull  * 128 * 2);
  _Float16* w42 = (_Float16*)alloc(512ull * 64  * 2);

  auto prep = [&](const float* W, _Float16* wt, int Kin, int Kpad, int Cout) {
    const int n = Kpad * Cout;
    prep_wt_kernel<<<(n + 255) / 256, 256, 0, stream>>>(W, wt, Kin, Kpad, Cout);
  };
  auto zero = [&](float* ptr, long long n) {
    fill_kernel<<<(int)((n + 255) / 256), 256, 0, stream>>>(ptr, (int)n);
  };

  // weight prep (f32 -> transposed padded f16)
  prep(c1W1, w11, 12, 32, 64);
  prep(c1W2, w12, 64, 64, 64);
  prep(c1W3, w13, 64, 64, 64);
  prep(c2W1, w21, 128, 128, 64);
  prep(c2W2, w22, 64, 64, 64);
  prep(c3W1, w31, 128, 128, 64);
  prep(c3W2, w32, 64, 64, 64);
  prep(c4W1, w41, 128, 128, 64);
  prep(c4W2, w42, 64, 64, 512);

  // input concat [pos|normal]
  concat_kernel<<<(16 * 2048 + 255) / 256, 256, 0, stream>>>(pos, nrm, x0, 16 * 2048);

  // ---- stage 1: N=2048, C=6 ----
  knn_kernel<6><<<dim3(2048 / 128, BATCH), 128, 0, stream>>>(x0, nbr, 2048);
  zero(x1, 16ll * 2048 * 64);
  edge_conv_kernel<32, 3, 64><<<dim3(2048 / NODE_BLK, BATCH), 128, 0, stream>>>(
      x0, nbr, w11, c1b1, w12, c1b2, w13, c1b3, x1, 2048, 6);
  gmax_kernel<<<dim3(1, BATCH), 256, 0, stream>>>(x1, f704, 2048, 64, 0);
  topk_pool_kernel<<<BATCH, 256, 0, stream>>>(x1, p1, x1p, 2048);

  // ---- stage 2: N=1024, C=64 ----
  knn_kernel<64><<<dim3(1024 / 128, BATCH), 128, 0, stream>>>(x1p, nbr, 1024);
  zero(x2, 16ll * 1024 * 64);
  edge_conv_kernel<128, 2, 64><<<dim3(1024 / NODE_BLK, BATCH), 128, 0, stream>>>(
      x1p, nbr, w21, c2b1, w22, c2b2, nullptr, nullptr, x2, 1024, 64);
  gmax_kernel<<<dim3(1, BATCH), 256, 0, stream>>>(x2, f704, 1024, 64, 64);
  topk_pool_kernel<<<BATCH, 256, 0, stream>>>(x2, p2, x2p, 1024);

  // ---- stage 3: N=512, C=64 ----
  knn_kernel<64><<<dim3(512 / 128, BATCH), 128, 0, stream>>>(x2p, nbr, 512);
  zero(x3, 16ll * 512 * 64);
  edge_conv_kernel<128, 2, 64><<<dim3(512 / NODE_BLK, BATCH), 128, 0, stream>>>(
      x2p, nbr, w31, c3b1, w32, c3b2, nullptr, nullptr, x3, 512, 64);
  gmax_kernel<<<dim3(1, BATCH), 256, 0, stream>>>(x3, f704, 512, 64, 128);
  topk_pool_kernel<<<BATCH, 256, 0, stream>>>(x3, p3, x3p, 512);

  // ---- stage 4: N=256, C=64 -> 512 ----
  knn_kernel<64><<<dim3(256 / 128, BATCH), 128, 0, stream>>>(x3p, nbr, 256);
  zero(x4, 16ll * 256 * 512);
  edge_conv_kernel<128, 2, 512><<<dim3(256 / NODE_BLK, BATCH), 128, 0, stream>>>(
      x3p, nbr, w41, c4b1, w42, c4b2, nullptr, nullptr, x4, 256, 64);
  gmax_kernel<<<dim3(2, BATCH), 256, 0, stream>>>(x4, f704, 256, 512, 192);

  // ---- classifier head ----
  linear_kernel<<<(16 * 512 + 255) / 256, 256, 0, stream>>>(f704, m1W, m1b, h512, 16, 704, 512, 1);
  linear_kernel<<<(16 * 256 + 255) / 256, 256, 0, stream>>>(h512, m2W, m2b, h256, 16, 512, 256, 1);
  linear_kernel<<<(16 * 40 + 255) / 256, 256, 0, stream>>>(h256, m3W, m3b, lgt, 16, 256, 40, 0);
  logsoftmax_kernel<<<1, 32, 0, stream>>>(lgt, dout);
}
